// GCNModel_16475494547624
// MI455X (gfx1250) — matrix operand order repaired and checked by reading, not compiled, not run on hardware
//
#include <hip/hip_runtime.h>
#include <hip/hip_bf16.h>

typedef __attribute__((ext_vector_type(16))) __bf16 v16bf;
typedef __attribute__((ext_vector_type(8)))  __bf16 v8bf;
typedef __attribute__((ext_vector_type(4)))  __bf16 v4bf;
typedef __attribute__((ext_vector_type(8)))  float  v8f;

__device__ __forceinline__ __bf16 f2bf(float f) {
  union { float f; unsigned u; } x; x.f = f;
  unsigned r = (x.u + 0x7FFFu + ((x.u >> 16) & 1u)) >> 16;   // round-to-nearest-even
  union { unsigned short s; __bf16 b; } y; y.s = (unsigned short)r;
  return y.b;
}
// HW packed converter: 2x f32 -> packed bf16 (RNE), one VOP3 per pair
__device__ __forceinline__ unsigned cvt_pk_bf16(float a, float b) {
  unsigned r;
  asm("v_cvt_pk_bf16_f32 %0, %1, %2" : "=v"(r) : "v"(a), "v"(b));
  return r;
}
__device__ __forceinline__ v4bf f2bf4(float4 v) {
  union { v4bf v4; unsigned u[2]; } u;
  u.u[0] = cvt_pk_bf16(v.x, v.y);
  u.u[1] = cvt_pk_bf16(v.z, v.w);
  return u.v4;
}
// unpack one 32-bit word holding two bf16
__device__ __forceinline__ float bflo(unsigned u) {
  union { unsigned u; float f; } y; y.u = u << 16; return y.f;
}
__device__ __forceinline__ float bfhi(unsigned u) {
  union { unsigned u; float f; } y; y.u = u & 0xFFFF0000u; return y.f;
}

// async global->LDS copy of 16 bytes per lane (CDNA5 async path, ASYNCcnt)
__device__ __forceinline__ void async_b128(unsigned lds_off, const float* gptr) {
  asm volatile("global_load_async_to_lds_b128 %0, %1, off"
               :: "v"(lds_off), "v"(gptr) : "memory");
}
__device__ __forceinline__ void wait_async0() {
  asm volatile("s_wait_asynccnt 0" ::: "memory");
}
__device__ __forceinline__ void wait_ds0() {
  asm volatile("s_wait_dscnt 0" ::: "memory");
}

// ---------------------------------------------------------------------------
// Degree / normalization kernels
// ---------------------------------------------------------------------------
__global__ void k_deg_init(float* __restrict__ deg, int n) {
  int i = blockIdx.x * blockDim.x + threadIdx.x;
  if (i < n) deg[i] = 1.0f;                       // self-loop contributes 1
}
__global__ void k_deg_acc(const long long* __restrict__ edst, float* __restrict__ deg, int E) {
  int i = blockIdx.x * blockDim.x + threadIdx.x;
  if (i < E) atomicAdd(&deg[(int)edst[i]], 1.0f);
}
__global__ void k_rsqrt(float* __restrict__ d, int n) {
  int i = blockIdx.x * blockDim.x + threadIdx.x;
  if (i < n) d[i] = rsqrtf(d[i]);
}

// agg[n] = xw[n] * dis[n]^2  (self-loop term)
__global__ void k_selfloop(const float* __restrict__ xw, const float* __restrict__ dis,
                           float* __restrict__ agg, int n) {
  long long i = (long long)blockIdx.x * blockDim.x + threadIdx.x;
  if (i < (long long)n * 16) {
    int node = (int)(i >> 4), q = (int)(i & 15);
    float c = dis[node]; c *= c;
    float4 v = ((const float4*)xw)[(size_t)node * 16 + q];
    v.x *= c; v.y *= c; v.z *= c; v.w *= c;
    ((float4*)agg)[(size_t)node * 16 + q] = v;
  }
}

// agg[dst] += xw[src] * dis[src]*dis[dst] for every edge (float4 gather + f32 atomics)
__global__ void k_scatter(const long long* __restrict__ esrc, const long long* __restrict__ edst,
                          const float* __restrict__ xw, const float* __restrict__ dis,
                          float* __restrict__ agg, int E) {
  long long i = (long long)blockIdx.x * blockDim.x + threadIdx.x;
  if (i < (long long)E * 16) {
    int e = (int)(i >> 4), q = (int)(i & 15);
    int s = (int)esrc[e], d = (int)edst[e];
    float c = dis[s] * dis[d];
    float4 v = ((const float4*)xw)[(size_t)s * 16 + q];
    float* a = agg + (size_t)d * 64 + q * 4;
    atomicAdd(a + 0, v.x * c); atomicAdd(a + 1, v.y * c);
    atomicAdd(a + 2, v.z * c); atomicAdd(a + 3, v.w * c);
  }
}

__global__ void k_bias_act(float* __restrict__ h, const float* __restrict__ b, int n, int relu) {
  long long i = (long long)blockIdx.x * blockDim.x + threadIdx.x;
  if (i < (long long)n * 64) {
    int f = (int)(i & 63);
    float v = h[i] + b[f];
    h[i] = relu ? fmaxf(v, 0.0f) : v;
  }
}

// ---------------------------------------------------------------------------
// Node-level GEMM: Y[N,64] = X[N,K] @ W[K,64]   (bf16 WMMA, f32 accumulate)
// One wave per 16-row tile; A rows arrive via async global->LDS (f32), software
// pipelined one tile ahead; W pre-swizzled in LDS into B-fragment order.
// ---------------------------------------------------------------------------
template <int K>
__global__ __launch_bounds__(128) void node_gemm(const float* __restrict__ X,
                                                 const float* __restrict__ W,
                                                 float* __restrict__ Y, int nrows) {
  constexpr int KC = K / 32;
  __shared__ __attribute__((aligned(32))) __bf16 wswz[KC * 4 * 512];   // [kc][ntile][lane][16]
  __shared__ __attribute__((aligned(16))) float  atile[4][16 * K];     // f32 A tiles (async dst)
  const int lane = threadIdx.x & 31;
  const int wave = threadIdx.x >> 5;
  const int m = lane & 15, h = lane >> 4;

  const int ntiles = (nrows + 15) >> 4;
  const int stride = gridDim.x * 4;
  int tile = blockIdx.x * 4 + wave;

  // lane gathers half a row (K/2 f32) for its tile, async into LDS
  auto issue = [&](int t) {
    int row = t * 16 + m; if (row >= nrows) row = nrows - 1;
    const float* gp = X + (size_t)row * K + h * (K / 2);
    unsigned lo = (unsigned)(unsigned long long)&atile[wave][m * K + h * (K / 2)];
#pragma unroll
    for (int i = 0; i < K / 8; ++i) async_b128(lo + i * 16, gp + i * 4);
  };
  if (tile < ntiles) issue(tile);   // prologue overlaps with weight staging below

  // stage W swizzled: element e of lane l's fragment for (kc=c, ntile=t) is
  // W[c*32 + 8*(l>>4) + (e&7) + 16*(e>>3)][t*16 + (l&15)]   (16-bit B layout, CDNA5 ISA)
  for (int idx = threadIdx.x; idx < KC * 4 * 512; idx += blockDim.x) {
    int e = idx & 15, l = (idx >> 4) & 31, f = idx >> 9;
    int t = f & 3, c = f >> 2;
    int k = c * 32 + 8 * (l >> 4) + (e & 7) + 16 * (e >> 3);
    wswz[idx] = f2bf(W[k * 64 + t * 16 + (l & 15)]);
  }
  __syncthreads();

  for (; tile < ntiles; tile += stride) {
    wait_async0();   // this tile's A rows have landed in LDS

    // A fragments: lane holds row (lane&15); K pattern {0..7,16..23}+8*(lane>>4)
    v16bf af[KC];
    const float* bp = &atile[wave][m * K];
#pragma unroll
    for (int c = 0; c < KC; ++c) {
      union { v16bf v; v4bf q[4]; } u;
#pragma unroll
      for (int g = 0; g < 2; ++g) {
        const float4* p = (const float4*)(bp + c * 32 + 16 * g + 8 * h);
        u.q[2 * g + 0] = f2bf4(p[0]);
        u.q[2 * g + 1] = f2bf4(p[1]);
      }
      af[c] = u.v;
    }
    wait_ds0();                                   // frags extracted; LDS reusable
    if (tile + stride < ntiles) issue(tile + stride);   // prefetch next tile

#pragma unroll
    for (int t = 0; t < 4; ++t) {
      v8f acc = {};
#pragma unroll
      for (int c = 0; c < KC; ++c) {
        union { v16bf v; v8bf s[2]; } u;
        const __bf16* p = &wswz[((c * 4 + t) * 32 + lane) * 16];
        u.s[0] = *(const v8bf*)p;
        u.s[1] = *(const v8bf*)(p + 8);
        acc = __builtin_amdgcn_wmma_f32_16x16x32_bf16(false, af[c], false, u.v,
                                                      (short)0, acc, false, false);
      }
      // D layout: VGPR j -> M = j + 8*(lane>>4), N = lane&15
#pragma unroll
      for (int j = 0; j < 8; ++j) {
        int rr = tile * 16 + j + 8 * h;
        if (rr < nrows) Y[(size_t)rr * 64 + t * 16 + (lane & 15)] = acc[j];
      }
    }
  }
}

// ---------------------------------------------------------------------------
// Edge classifier: out[E,2] = relu(concat(H[src],H[dst]) @ Wc1 + bc1) @ Wc2 + bc2
// One wave per 16-edge tile; endpoint rows gathered async global->LDS (L2
// resident), pipelined one tile ahead; 16 WMMAs; tiny head via VALU from LDS.
// ---------------------------------------------------------------------------
__global__ __launch_bounds__(128) void edge_mlp(const long long* __restrict__ esrc,
                                                const long long* __restrict__ edst,
                                                const float* __restrict__ H,
                                                const float* __restrict__ Wc1,
                                                const float* __restrict__ bc1,
                                                const float* __restrict__ Wc2,
                                                const float* __restrict__ bc2,
                                                float* __restrict__ out, int E) {
  __shared__ __attribute__((aligned(32))) __bf16 wswz[4 * 4 * 512];    // Wc1 swizzled, 16 KB
  __shared__ __attribute__((aligned(16))) float  atile[4][16 * 128];   // f32 A tiles, 32 KB
  __shared__ __attribute__((aligned(16))) __bf16 ztile[4][16 * 64];    // 8 KB
  __shared__ float wc2s[128], bc1s[64], bc2s[2];
  const int lane = threadIdx.x & 31;
  const int wave = threadIdx.x >> 5;
  const int m = lane & 15, h = lane >> 4;

  const int ntiles = (E + 15) >> 4;
  const int stride = gridDim.x * 4;
  int tile = blockIdx.x * 4 + wave;

  // lane gathers one endpoint embedding (64 f32) for edge (tile*16 + m)
  auto issue = [&](int t) {
    long long eg = (long long)t * 16 + m;
    if (eg >= E) eg = (long long)E - 1;
    int node = (int)(h ? edst[eg] : esrc[eg]);
    const float* gp = H + (size_t)node * 64;
    unsigned lo = (unsigned)(unsigned long long)&atile[wave][m * 128 + h * 64];
#pragma unroll
    for (int i = 0; i < 16; ++i) async_b128(lo + i * 16, gp + i * 4);
  };
  if (tile < ntiles) issue(tile);   // prologue overlaps with weight staging

  for (int idx = threadIdx.x; idx < 8192; idx += blockDim.x) {
    int e = idx & 15, l = (idx >> 4) & 31, f = idx >> 9;
    int t = f & 3, c = f >> 2;
    int k = c * 32 + 8 * (l >> 4) + (e & 7) + 16 * (e >> 3);
    wswz[idx] = f2bf(Wc1[k * 64 + t * 16 + (l & 15)]);
  }
  if (threadIdx.x < 128) wc2s[threadIdx.x] = Wc2[threadIdx.x];
  if (threadIdx.x < 64)  bc1s[threadIdx.x] = bc1[threadIdx.x];
  if (threadIdx.x < 2)   bc2s[threadIdx.x] = bc2[threadIdx.x];
  __syncthreads();

  for (; tile < ntiles; tile += stride) {
    wait_async0();   // gathered endpoint rows have landed

    v16bf af[4];
    const float* bp = &atile[wave][m * 128];
#pragma unroll
    for (int c = 0; c < 4; ++c) {
      union { v16bf v; v4bf q[4]; } u;
#pragma unroll
      for (int g = 0; g < 2; ++g) {
        const float4* p = (const float4*)(bp + c * 32 + 16 * g + 8 * h);
        u.q[2 * g + 0] = f2bf4(p[0]);
        u.q[2 * g + 1] = f2bf4(p[1]);
      }
      af[c] = u.v;
    }
    wait_ds0();
    if (tile + stride < ntiles) issue(tile + stride);   // prefetch next 16 edges

#pragma unroll
    for (int t = 0; t < 4; ++t) {
      v8f acc = {};
#pragma unroll
      for (int c = 0; c < 4; ++c) {
        union { v16bf v; v8bf s[2]; } u;
        const __bf16* p = &wswz[((c * 4 + t) * 32 + lane) * 16];
        u.s[0] = *(const v8bf*)p;
        u.s[1] = *(const v8bf*)(p + 8);
        acc = __builtin_amdgcn_wmma_f32_16x16x32_bf16(false, af[c], false, u.v,
                                                      (short)0, acc, false, false);
      }
#pragma unroll
      for (int j = 0; j < 8; ++j) {
        int rr = j + 8 * h;
        float z = acc[j] + bc1s[t * 16 + (lane & 15)];
        ztile[wave][rr * 64 + t * 16 + (lane & 15)] = f2bf(fmaxf(z, 0.0f));
      }
    }
    wait_ds0();

    // head: lane -> (edge = lane>>1, class = lane&1); 64-term dot, z row read
    // as 8x ds_load_b128 (8 bf16 per load) and unpacked in registers
    int e2 = lane >> 1, cls = lane & 1;
    const uint4* zr = (const uint4*)&ztile[wave][e2 * 64];
    float acc2 = bc2s[cls];
#pragma unroll
    for (int q = 0; q < 8; ++q) {
      uint4 w = zr[q];
      const float* wp = &wc2s[(q * 8) * 2 + cls];
      acc2 = fmaf(bflo(w.x), wp[0],  acc2);
      acc2 = fmaf(bfhi(w.x), wp[2],  acc2);
      acc2 = fmaf(bflo(w.y), wp[4],  acc2);
      acc2 = fmaf(bfhi(w.y), wp[6],  acc2);
      acc2 = fmaf(bflo(w.z), wp[8],  acc2);
      acc2 = fmaf(bfhi(w.z), wp[10], acc2);
      acc2 = fmaf(bflo(w.w), wp[12], acc2);
      acc2 = fmaf(bfhi(w.w), wp[14], acc2);
    }
    long long eo = (long long)tile * 16 + e2;
    if (eo < E) out[eo * 2 + cls] = acc2;
  }
}

// ---------------------------------------------------------------------------
extern "C" void kernel_launch(void* const* d_in, const int* in_sizes, int n_in,
                              void* d_out, int out_size, void* d_ws, size_t ws_size,
                              hipStream_t stream) {
  (void)n_in; (void)out_size; (void)ws_size;
  const float*     x    = (const float*)d_in[0];
  const long long* ei   = (const long long*)d_in[1];
  const float*     W1   = (const float*)d_in[2];
  const float*     b1   = (const float*)d_in[3];
  const float*     W2   = (const float*)d_in[4];
  const float*     b2   = (const float*)d_in[5];
  const float*     Wc1  = (const float*)d_in[6];
  const float*     bc1  = (const float*)d_in[7];
  const float*     Wc2  = (const float*)d_in[8];
  const float*     bc2  = (const float*)d_in[9];
  float* out = (float*)d_out;

  const int N = in_sizes[0] / 128;
  const int E = in_sizes[1] / 2;
  const long long* esrc = ei;
  const long long* edst = ei + E;

  char* ws = (char*)d_ws;
  float* buf0 = (float*)ws;                            // [N,64]
  float* buf1 = (float*)(ws + (size_t)N * 64 * 4);     // [N,64]
  float* dis  = (float*)(ws + (size_t)N * 64 * 8);     // [N] deg -> rsqrt(deg)

  const int T = 256;
  auto cdiv = [](long long a, long long b) { return (int)((a + b - 1) / b); };

  // normalization
  k_deg_init<<<cdiv(N, T), T, 0, stream>>>(dis, N);
  k_deg_acc<<<cdiv(E, T), T, 0, stream>>>(edst, dis, E);
  k_rsqrt<<<cdiv(N, T), T, 0, stream>>>(dis, N);

  const int tilesN  = (N + 15) / 16;
  const int blocksN = (tilesN + 3) / 4;

  // layer 1
  node_gemm<128><<<blocksN, 128, 0, stream>>>(x, W1, buf0, N);
  k_selfloop<<<cdiv((long long)N * 16, T), T, 0, stream>>>(buf0, dis, buf1, N);
  k_scatter<<<cdiv((long long)E * 16, T), T, 0, stream>>>(esrc, edst, buf0, dis, buf1, E);
  k_bias_act<<<cdiv((long long)N * 64, T), T, 0, stream>>>(buf1, b1, N, 1);   // h1 = buf1

  // layer 2
  node_gemm<64><<<blocksN, 128, 0, stream>>>(buf1, W2, buf0, N);
  k_selfloop<<<cdiv((long long)N * 16, T), T, 0, stream>>>(buf0, dis, buf1, N);
  k_scatter<<<cdiv((long long)E * 16, T), T, 0, stream>>>(esrc, edst, buf0, dis, buf1, E);
  k_bias_act<<<cdiv((long long)N * 64, T), T, 0, stream>>>(buf1, b2, N, 0);   // h2 = buf1

  // edge classifier (dominant: 26 GFLOP via WMMA, async-pipelined L2 gathers)
  const int tilesE  = (E + 15) / 16;
  const int blocksE = (tilesE + 3) / 4;
  edge_mlp<<<blocksE, 128, 0, stream>>>(esrc, edst, buf1, Wc1, bc1, Wc2, bc2, out, E);
}
